// DynamicMLP_53541062312307
// MI455X (gfx1250) — matrix operand order repaired and checked by reading, not compiled, not run on hardware
//
#include <hip/hip_runtime.h>
#include <hip/hip_bf16.h>

typedef __attribute__((ext_vector_type(16))) _Float16 v16h;
typedef __attribute__((ext_vector_type(8)))  _Float16 v8h;
typedef __attribute__((ext_vector_type(8)))  float    v8f;

// ---------------------------------------------------------------------------
// f32 -> f16 weight conversion
// ---------------------------------------------------------------------------
__global__ void snn_f32_to_f16(const float* __restrict__ src,
                               _Float16* __restrict__ dst, int n) {
    int i = blockIdx.x * blockDim.x + threadIdx.x;
    if (i < n) dst[i] = (_Float16)src[i];
}

// ---------------------------------------------------------------------------
// Pack input spikes [B, D, T] (f32) -> [T, B, D] (f16), writes coalesced.
// ---------------------------------------------------------------------------
__global__ void snn_pack_in(const float* __restrict__ in,
                            _Float16* __restrict__ S,
                            int Bt, int D, int T) {
    size_t gid = (size_t)blockIdx.x * blockDim.x + threadIdx.x;
    size_t total = (size_t)Bt * D * T;
    if (gid >= total) return;
    int d = (int)(gid % D);
    size_t bt = gid / D;          // = t*Bt + b
    int b = (int)(bt % Bt);
    int t = (int)(bt / Bt);
    S[gid] = (_Float16)in[((size_t)b * D + d) * T + t];
}

// ---------------------------------------------------------------------------
// WMMA fragment helpers.
// A 16x32 f16 frag: lane = row (m%16); lanes 0-15 hold K {0..7,16..23},
//                   lanes 16-31 hold K {8..15,24..31}  -> two 16B loads/lane
// B 32x16 f16 frag: lane = col (n%16); lanes 0-15 hold K {0..7,16..23},
//                   lanes 16-31 hold K {8..15,24..31}  -> two 16B loads/lane
// ---------------------------------------------------------------------------
__device__ __forceinline__ void load_tile(const _Float16* __restrict__ Arow0,
                                          const _Float16* __restrict__ Arow1,
                                          const _Float16* __restrict__ Bbase,
                                          size_t bstride, int k0,
                                          v16h& a0, v16h& a1, v16h (&b)[4]) {
    union { v16h v; v8h h[2]; } u;
    u.h[0] = *(const v8h*)(Arow0 + k0);
    u.h[1] = *(const v8h*)(Arow0 + k0 + 16);
    a0 = u.v;
    u.h[0] = *(const v8h*)(Arow1 + k0);
    u.h[1] = *(const v8h*)(Arow1 + k0 + 16);
    a1 = u.v;
#pragma unroll
    for (int j = 0; j < 4; ++j) {
        u.h[0] = *(const v8h*)(Bbase + (size_t)j * bstride + k0);
        u.h[1] = *(const v8h*)(Bbase + (size_t)j * bstride + k0 + 16);
        b[j] = u.v;
    }
}

__device__ __forceinline__ void mma_tile(const v16h& a0, const v16h& a1,
                                         const v16h (&b)[4],
                                         v8f (&acc0)[4], v8f (&acc1)[4]) {
#pragma unroll
    for (int j = 0; j < 4; ++j) {
        acc0[j] = __builtin_amdgcn_wmma_f32_16x16x32_f16(
            false, a0, false, b[j], (short)0, acc0[j], false, false);
        acc1[j] = __builtin_amdgcn_wmma_f32_16x16x32_f16(
            false, a1, false, b[j], (short)0, acc1[j], false, false);
    }
}

// ---------------------------------------------------------------------------
// WMMA GEMM:  X[t] = S[t] * W^T + bias
//   S: [T][Bt][K] f16 (spikes, K contiguous)
//   W: [N][K]     f16 (row n of W == column n of the K x N WMMA B matrix)
//   X: [T][Bt][N] f32
// Block = 256 thr = 8 waves, 2(M) x 4(N); wave tile = 32 x 64
// (2 A-frags x 4 B-frags = 8 WMMAs / K-step, double-buffered K pipeline).
// Block tile = 64(M) x 256(N). Grid = (N/256, Bt/64, T). K % 64 == 0.
// ---------------------------------------------------------------------------
__global__ __launch_bounds__(256)
void snn_gemm_wmma(const _Float16* __restrict__ S,
                   const _Float16* __restrict__ W,
                   const float* __restrict__ bias,
                   float* __restrict__ X,
                   int Bt, int K, int N) {
    const int lane = threadIdx.x & 31;
    const int wave = threadIdx.x >> 5;
    const int wm   = wave & 1;              // 2 waves along M (32 rows each)
    const int wn   = wave >> 1;             // 4 waves along N (64 cols each)
    const int t    = blockIdx.z;

    const int row0 = blockIdx.y * 64 + wm * 32;
    const int col0 = blockIdx.x * 256 + wn * 64;

    const int mlane = lane & 15;
    const int aoff  = (lane & 16) ? 8 : 0;
    const _Float16* Sbase = S + (size_t)t * Bt * K;
    const _Float16* Arow0 = Sbase + (size_t)(row0 + mlane) * K + aoff;
    const _Float16* Arow1 = Sbase + (size_t)(row0 + 16 + mlane) * K + aoff;

    const int nlane = col0 + mlane;
    const int kb    = (lane & 16) ? 8 : 0;
    const _Float16* Bbase = W + (size_t)nlane * K + kb;
    const size_t bstride  = (size_t)16 * K;     // 16 columns apart

    v8f acc0[4] = {};
    v8f acc1[4] = {};

    v16h a00, a01, b0[4];
    v16h a10, a11, b1[4];

    load_tile(Arow0, Arow1, Bbase, bstride, 0, a00, a01, b0);
    for (int k0 = 0; k0 < K; k0 += 64) {
        // prefetch A stream well ahead into near caches
        if (k0 + 512 < K) {
            __builtin_prefetch(Arow0 + k0 + 512, 0, 3);
            __builtin_prefetch(Arow1 + k0 + 512, 0, 3);
        }
        load_tile(Arow0, Arow1, Bbase, bstride, k0 + 32, a10, a11, b1);
        mma_tile(a00, a01, b0, acc0, acc1);
        if (k0 + 64 < K)
            load_tile(Arow0, Arow1, Bbase, bstride, k0 + 64, a00, a01, b0);
        mma_tile(a10, a11, b1, acc0, acc1);
    }

    // C/D layout: VGPR r, lanes 0-15 -> M=r, lanes 16-31 -> M=8+r, N=lane%16
    const int mrow = row0 + ((lane & 16) ? 8 : 0);
    float* Xt = X + (size_t)t * Bt * N;
#pragma unroll
    for (int j = 0; j < 4; ++j) {
        const int n = nlane + j * 16;
        const float bb = bias[n];
#pragma unroll
        for (int r = 0; r < 8; ++r) {
            Xt[(size_t)(mrow + r) * N + n]      = acc0[j][r] + bb;
            Xt[(size_t)(mrow + 16 + r) * N + n] = acc1[j][r] + bb;
        }
    }
}

// ---------------------------------------------------------------------------
// LIF neuron recurrence, one thread per (b, h), serial over T.
//   X:   [T][BH] f32 pre-activations
//   Sout:[T][BH] f16 spikes (optional)
//   out: [BH]    f32 spike rate = sum_t s / T (optional, final layer)
// ---------------------------------------------------------------------------
__global__ void snn_lif(const float* __restrict__ X,
                        _Float16* __restrict__ Sout,
                        float* __restrict__ out, int BH, int T) {
    int i = blockIdx.x * blockDim.x + threadIdx.x;
    if (i >= BH) return;
    float c = 0.f, v = 0.f, u = 0.f, s = 0.f, sum = 0.f;
    for (int t = 0; t < T; ++t) {
        float x = X[(size_t)t * BH + i];
        c = c * 0.5f + x;                       // NEURON_CDECAY
        v = v * (1.0f - s) + 0.021f * s;        // THETA_r
        u = u + s * 0.132f;                     // THETA_s
        float dv = v * v - v - u + c;
        float du = -0.172f * v + 0.529f * u;    // THETA_v, THETA_u
        v += dv;
        u += du;
        s = (v > 0.5f) ? 1.0f : 0.0f;           // NEURON_VTH
        if (Sout) Sout[(size_t)t * BH + i] = (_Float16)s;
        sum += s;
    }
    if (out) out[i] = sum * (1.0f / T);
}

// ---------------------------------------------------------------------------
// Launch
// ---------------------------------------------------------------------------
extern "C" void kernel_launch(void* const* d_in, const int* in_sizes, int n_in,
                              void* d_out, int out_size, void* d_ws, size_t ws_size,
                              hipStream_t stream) {
    const float* in = (const float*)d_in[0];
    const float* W1 = (const float*)d_in[1];
    const float* b1 = (const float*)d_in[2];
    const float* W2 = (const float*)d_in[3];
    const float* b2 = (const float*)d_in[4];
    const float* Wo = (const float*)d_in[5];
    const float* bo = (const float*)d_in[6];

    const int B = 512, Din = 2048, H1 = 2048, H2 = 2048, Dout = 1024, T = 16;

    char* ws = (char*)d_ws;
    _Float16* W1h  = (_Float16*)(ws);                       // 8 MB
    _Float16* W2h  = (_Float16*)(ws + (size_t)(8u  << 20)); // 8 MB
    _Float16* Woh  = (_Float16*)(ws + (size_t)(16u << 20)); // 4 MB
    _Float16* Sbuf = (_Float16*)(ws + (size_t)(20u << 20)); // 32 MB, T*B*2048 f16
    float*    Xbuf = (float*)   (ws + (size_t)(52u << 20)); // 64 MB, T*B*2048 f32

    // Weight conversion f32 -> f16
    snn_f32_to_f16<<<(H1 * Din + 255) / 256, 256, 0, stream>>>(W1, W1h, H1 * Din);
    snn_f32_to_f16<<<(H2 * H1 + 255) / 256, 256, 0, stream>>>(W2, W2h, H2 * H1);
    snn_f32_to_f16<<<(Dout * H2 + 255) / 256, 256, 0, stream>>>(Wo, Woh, Dout * H2);

    // Pack input spikes [B,Din,T] -> f16 [T,B,Din]
    {
        size_t total = (size_t)B * Din * T;
        snn_pack_in<<<(unsigned)((total + 255) / 256), 256, 0, stream>>>(
            in, Sbuf, B, Din, T);
    }

    // Layer 1
    snn_gemm_wmma<<<dim3(H1 / 256, B / 64, T), 256, 0, stream>>>(
        Sbuf, W1h, b1, Xbuf, B, Din, H1);
    snn_lif<<<(B * H1 + 255) / 256, 256, 0, stream>>>(Xbuf, Sbuf, nullptr, B * H1, T);

    // Layer 2
    snn_gemm_wmma<<<dim3(H2 / 256, B / 64, T), 256, 0, stream>>>(
        Sbuf, W2h, b2, Xbuf, B, H1, H2);
    snn_lif<<<(B * H2 + 255) / 256, 256, 0, stream>>>(Xbuf, Sbuf, nullptr, B * H2, T);

    // Output layer + rate decode
    snn_gemm_wmma<<<dim3(Dout / 256, B / 64, T), 256, 0, stream>>>(
        Sbuf, Woh, bo, Xbuf, B, H2, Dout);
    snn_lif<<<(B * Dout + 255) / 256, 256, 0, stream>>>(
        Xbuf, nullptr, (float*)d_out, B * Dout, T);
}
